// biGRU_72550587564390
// MI455X (gfx1250) — matrix operand order repaired and checked by reading, not compiled, not run on hardware
//
#include <hip/hip_runtime.h>

#define E_DIM   300
#define EPAD    320
#define H_DIM   512
#define OUT_DIM 9
#define B_SZ    64
#define S_LEN   256
#define ROWS    (B_SZ * S_LEN)   // 16384

typedef __attribute__((ext_vector_type(16))) __bf16       v16bf;
typedef __attribute__((ext_vector_type(8)))  float        v8f;
typedef __attribute__((ext_vector_type(4)))  unsigned int u32x4;

union Frag { u32x4 q[2]; v16bf v; };

__device__ __forceinline__ unsigned short f2bf(float f) {
  unsigned int u = __float_as_uint(f);
  u += 0x7FFFu + ((u >> 16) & 1u);          // round-to-nearest-even
  return (unsigned short)(u >> 16);
}
__device__ __forceinline__ float bf2f(unsigned short s) {
  return __uint_as_float(((unsigned int)s) << 16);
}
__device__ __forceinline__ float sigmoidf_(float x) {
  return 1.0f / (1.0f + __expf(-x));
}
__device__ __forceinline__ v8f zero8() {
  v8f z;
  #pragma unroll
  for (int i = 0; i < 8; ++i) z[i] = 0.0f;
  return z;
}

// ---------------- K0: embedding gather -> bf16, E padded 300->320 -------------
__global__ __launch_bounds__(256) void biGRU_embed_kernel(
    const int* __restrict__ x, const float* __restrict__ emb,
    unsigned short* __restrict__ xe) {
  int idx = blockIdx.x * 256 + threadIdx.x;            // [16384][320]
  if (idx >= ROWS * EPAD) return;
  int e   = idx % EPAD;
  int row = idx / EPAD;                                // row = t*64 + b
  int t   = row >> 6;
  int b   = row & 63;
  float v = 0.0f;
  if (e < E_DIM) {
    int tok = x[b * S_LEN + t];
    v = emb[(size_t)tok * E_DIM + e];
  }
  xe[idx] = f2bf(v);
}

// ---------------- K1: weight split + convert to bf16 -------------------------
// Wx[g][h][e]  (e padded to 320, zero-filled) = W[g][h][512+e]
// Wh[g][h][k]                                 = W[g][h][k]
__global__ __launch_bounds__(256) void biGRU_convw_kernel(
    const float* __restrict__ W0, const float* __restrict__ W1,
    const float* __restrict__ W2, const float* __restrict__ W3,
    const float* __restrict__ W4, const float* __restrict__ W5,
    unsigned short* __restrict__ Wx, unsigned short* __restrict__ Wh) {
  const int NX = 6 * H_DIM * EPAD;                     // 983040
  const int NH = 6 * H_DIM * H_DIM;                    // 1572864
  int idx = blockIdx.x * 256 + threadIdx.x;
  if (idx >= NX + NH) return;
  if (idx < NX) {
    int e = idx % EPAD; int rest = idx / EPAD;
    int hrow = rest % H_DIM; int g = rest / H_DIM;
    const float* W = (g == 0) ? W0 : (g == 1) ? W1 : (g == 2) ? W2
                   : (g == 3) ? W3 : (g == 4) ? W4 : W5;
    float v = (e < E_DIM) ? W[(size_t)hrow * (H_DIM + E_DIM) + H_DIM + e] : 0.0f;
    Wx[idx] = f2bf(v);
  } else {
    int j = idx - NX;
    int k = j % H_DIM; int rest = j / H_DIM;
    int hrow = rest % H_DIM; int g = rest / H_DIM;
    const float* W = (g == 0) ? W0 : (g == 1) ? W1 : (g == 2) ? W2
                   : (g == 3) ? W3 : (g == 4) ? W4 : W5;
    Wh[j] = f2bf(W[(size_t)hrow * (H_DIM + E_DIM) + k]);
  }
}

// ---------------- K2: Xin[g] = xe @ Wx[g].T  (bf16 WMMA, f32 accum) ----------
// One wave per 16x16 C tile. jobs = 6 * 1024 Mtiles * 32 Ntiles = 196608.
__global__ __launch_bounds__(256) void biGRU_xin_gemm_kernel(
    const unsigned short* __restrict__ xe, const unsigned short* __restrict__ Wx,
    unsigned short* __restrict__ Xin) {
  const int lane = threadIdx.x & 31;
  const int wave = threadIdx.x >> 5;
  const int job  = blockIdx.x * 8 + wave;
  const int g    = job >> 15;                          // / (1024*32)
  const int rem  = job & 32767;
  const int mt   = rem >> 5;
  const int nt   = rem & 31;
  const int h    = lane >> 4;
  const int mrow = (mt << 4) + (lane & 15);
  const int ncol = (nt << 4) + (lane & 15);

  const unsigned short* arow = xe + (size_t)mrow * EPAD;
  const unsigned short* brow = Wx + ((size_t)g * H_DIM + ncol) * EPAD;

  v8f acc = zero8();
  #pragma unroll
  for (int kc = 0; kc < EPAD; kc += 32) {
    Frag af, bf;
    af.q[0] = *(const u32x4*)(arow + kc + 8 * h);
    af.q[1] = *(const u32x4*)(arow + kc + 8 * h + 16);
    bf.q[0] = *(const u32x4*)(brow + kc + 8 * h);
    bf.q[1] = *(const u32x4*)(brow + kc + 8 * h + 16);
    acc = __builtin_amdgcn_wmma_f32_16x16x32_bf16(false, af.v, false, bf.v,
                                                  (short)0, acc, false, false);
  }
  unsigned short* out = Xin + ((size_t)g * ROWS + (mt << 4)) * H_DIM
                            + (nt << 4) + (lane & 15);
  #pragma unroll
  for (int v = 0; v < 8; ++v)
    out[(size_t)(v + 8 * h) * H_DIM] = f2bf(acc[v]);
}

// recurrent matmul helper: acc[mt] += A(lds) x Wh[col-row], K = 512
__device__ __forceinline__ void gru_mm4(const unsigned short* __restrict__ Wg,
                                        const unsigned short* ldsA,
                                        int lane, int h, int col, v8f acc[4]) {
  const unsigned short* brow = Wg + (size_t)col * H_DIM;   // B(k,n)=Wh[n][k]
  for (int kc = 0; kc < H_DIM; kc += 32) {
    Frag bf;
    bf.q[0] = *(const u32x4*)(brow + kc + 8 * h);
    bf.q[1] = *(const u32x4*)(brow + kc + 8 * h + 16);
    #pragma unroll
    for (int mt = 0; mt < 4; ++mt) {
      Frag af;
      const unsigned short* ar = ldsA + ((mt << 4) + (lane & 15)) * H_DIM + kc + 8 * h;
      af.q[0] = *(const u32x4*)(ar);
      af.q[1] = *(const u32x4*)(ar + 16);
      acc[mt] = __builtin_amdgcn_wmma_f32_16x16x32_bf16(false, af.v, false, bf.v,
                                                        (short)0, acc[mt], false, false);
    }
  }
}

// ---------------- K3: sequential recurrence, 1 workgroup per direction -------
__global__ __launch_bounds__(1024) void biGRU_seq_kernel(
    const unsigned short* __restrict__ Wh, const unsigned short* __restrict__ Xin,
    const float* __restrict__ bcf, const float* __restrict__ brf, const float* __restrict__ buf_,
    const float* __restrict__ bcb, const float* __restrict__ brb, const float* __restrict__ bub,
    float* __restrict__ cfout, float* __restrict__ cbout) {
  __shared__ unsigned short ldsC[B_SZ * H_DIM];        // 64 KB: c / r*c (bf16)

  const int dir  = blockIdx.x;                          // 0 fwd, 1 bwd
  const int lane = threadIdx.x & 31;
  const int w    = threadIdx.x >> 5;                    // wave == ntile (0..31)
  const int h    = lane >> 4;
  const int col  = (w << 4) + (lane & 15);

  const int gc = dir ? 3 : 0, gr = dir ? 4 : 1, gu = dir ? 5 : 2;
  const float* bc = dir ? bcb : bcf;
  const float* br = dir ? brb : brf;
  const float* bu = dir ? bub : buf_;
  float* cout = dir ? cbout : cfout;

  const unsigned short* WhR = Wh + (size_t)gr * H_DIM * H_DIM;
  const unsigned short* WhU = Wh + (size_t)gu * H_DIM * H_DIM;
  const unsigned short* WhC = Wh + (size_t)gc * H_DIM * H_DIM;

  for (int i = threadIdx.x; i < B_SZ * H_DIM; i += 1024) ldsC[i] = 0;
  v8f cReg[4];
  #pragma unroll
  for (int mt = 0; mt < 4; ++mt) cReg[mt] = zero8();

  const float bcv = bc[col], brv = br[col], buv = bu[col];
  __syncthreads();

  for (int step = 0; step < S_LEN; ++step) {
    const int t = dir ? (S_LEN - 1 - step) : step;
    const size_t xrow = (size_t)t * B_SZ;

    // sub-step 1: r,u recurrent matmuls (A = c, bf16 in LDS)
    v8f accR[4], accU[4];
    #pragma unroll
    for (int mt = 0; mt < 4; ++mt) { accR[mt] = zero8(); accU[mt] = zero8(); }
    gru_mm4(WhR, ldsC, lane, h, col, accR);
    gru_mm4(WhU, ldsC, lane, h, col, accU);
    __syncthreads();                                    // everyone done reading c

    // gate eval: r -> write r*c to LDS; u stays in registers
    v8f uReg[4];
    #pragma unroll
    for (int mt = 0; mt < 4; ++mt) {
      #pragma unroll
      for (int v = 0; v < 8; ++v) {
        const int brow = (mt << 4) + v + (h << 3);
        const size_t ro = (((size_t)gr * ROWS + xrow + brow) << 9) + col;
        const size_t uo = (((size_t)gu * ROWS + xrow + brow) << 9) + col;
        float r = sigmoidf_(accR[mt][v] + bf2f(Xin[ro]) + brv);
        uReg[mt][v] = sigmoidf_(accU[mt][v] + bf2f(Xin[uo]) + buv);
        ldsC[brow * H_DIM + col] = f2bf(r * cReg[mt][v]);
      }
    }
    __syncthreads();                                    // r*c visible

    // sub-step 2: candidate matmul (A = r*c in LDS)
    v8f accC[4];
    #pragma unroll
    for (int mt = 0; mt < 4; ++mt) accC[mt] = zero8();
    gru_mm4(WhC, ldsC, lane, h, col, accC);
    __syncthreads();                                    // done reading r*c

    // c_new = u*cc + (1-u)*c ; write bf16 to LDS + f32 to global
    #pragma unroll
    for (int mt = 0; mt < 4; ++mt) {
      #pragma unroll
      for (int v = 0; v < 8; ++v) {
        const int brow = (mt << 4) + v + (h << 3);
        const size_t co = (((size_t)gc * ROWS + xrow + brow) << 9) + col;
        float cc = tanhf(accC[mt][v] + bf2f(Xin[co]) + bcv);
        float u  = uReg[mt][v];
        float cn = u * cc + (1.0f - u) * cReg[mt][v];
        cReg[mt][v] = cn;
        ldsC[brow * H_DIM + col] = f2bf(cn);
        cout[((xrow + brow) << 9) + col] = cn;
      }
    }
    __syncthreads();                                    // new c visible
  }
}

// ---------------- K4: output projection [S*B,1024] x [1024,9] ----------------
__global__ __launch_bounds__(256) void biGRU_proj_kernel(
    const float* __restrict__ cf, const float* __restrict__ cb,
    const float* __restrict__ Wp, const float* __restrict__ bp,
    float* __restrict__ y) {
  int idx = blockIdx.x * 256 + threadIdx.x;             // b*S*9 + t*9 + o
  if (idx >= B_SZ * S_LEN * OUT_DIM) return;
  int o  = idx % OUT_DIM;
  int bt = idx / OUT_DIM;                               // bt = b*S + t
  int t  = bt % S_LEN;
  int b  = bt / S_LEN;
  const float* a1 = cf + ((size_t)t * B_SZ + b) * H_DIM;
  const float* a2 = cb + ((size_t)t * B_SZ + b) * H_DIM;
  const float* wr = Wp + (size_t)o * (2 * H_DIM);
  float s = bp[o];
  for (int k = 0; k < H_DIM; ++k)
    s += a1[k] * wr[k] + a2[k] * wr[k + H_DIM];
  y[idx] = s;
}

extern "C" void kernel_launch(void* const* d_in, const int* in_sizes, int n_in,
                              void* d_out, int out_size, void* d_ws, size_t ws_size,
                              hipStream_t stream) {
  const int*   x    = (const int*)  d_in[0];
  const float* emb  = (const float*)d_in[1];
  const float* Wcf  = (const float*)d_in[2];
  const float* bcf  = (const float*)d_in[3];
  const float* Wrf  = (const float*)d_in[4];
  const float* brf  = (const float*)d_in[5];
  const float* Wuf  = (const float*)d_in[6];
  const float* buf  = (const float*)d_in[7];
  const float* Wcb  = (const float*)d_in[8];
  const float* bcb  = (const float*)d_in[9];
  const float* Wrb  = (const float*)d_in[10];
  const float* brb  = (const float*)d_in[11];
  const float* Wub  = (const float*)d_in[12];
  const float* bub  = (const float*)d_in[13];
  const float* Wp   = (const float*)d_in[14];
  const float* bp   = (const float*)d_in[15];
  float* y = (float*)d_out;

  char* ws = (char*)d_ws;
  unsigned short* xe  = (unsigned short*)ws; ws += (size_t)ROWS * EPAD * 2;          // 10.5 MB
  unsigned short* Wx  = (unsigned short*)ws; ws += (size_t)6 * H_DIM * EPAD * 2;     //  2.0 MB
  unsigned short* Whb = (unsigned short*)ws; ws += (size_t)6 * H_DIM * H_DIM * 2;    //  3.1 MB
  unsigned short* Xin = (unsigned short*)ws; ws += (size_t)6 * ROWS * H_DIM * 2;     // 100.7 MB
  float* cfout = (float*)ws;                 ws += (size_t)ROWS * H_DIM * 4;         // 33.6 MB
  float* cbout = (float*)ws;                                                         // 33.6 MB

  biGRU_embed_kernel   <<<(ROWS * EPAD) / 256, 256, 0, stream>>>(x, emb, xe);
  biGRU_convw_kernel   <<<(6 * H_DIM * (EPAD + H_DIM)) / 256, 256, 0, stream>>>(
      Wcf, Wrf, Wuf, Wcb, Wrb, Wub, Wx, Whb);
  biGRU_xin_gemm_kernel<<<(6 * (ROWS / 16) * (H_DIM / 16)) / 8, 256, 0, stream>>>(
      xe, Wx, Xin);
  biGRU_seq_kernel     <<<2, 1024, 0, stream>>>(
      Whb, Xin, bcf, brf, buf, bcb, brb, bub, cfout, cbout);
  biGRU_proj_kernel    <<<(B_SZ * S_LEN * OUT_DIM + 255) / 256, 256, 0, stream>>>(
      cfout, cbout, Wp, bp, y);
}
